// GINModel_79517024518682
// MI455X (gfx1250) — compile-verified, matrix operand order checked by reading, not run on hardware
//
#include <hip/hip_runtime.h>
#include <hip/hip_bf16.h>

// ---------------------------------------------------------------------------
// CDNA5 (gfx1250) GIN pipeline.
//  - All GEMMs: v_wmma_f32_16x16x32_bf16 with inline-0 C init, fp32 accum;
//    bias added in the epilogue. A-fragments: 4x global_load_b128 per k-tile
//    + v_cvt_pk_bf16_f32. Weights pre-converted once per call into a
//    fragment-ready bf16 layout (WMMA B-layout baked in), staged in LDS and
//    read as ds_load_b128 pairs.
//  - DUAL epilogue: conv-layer GEMMs write h into both the JK concat buffer
//    and the next layer's scatter accumulator, eliminating strided re-copies.
//  - Scatter / pool: per-edge float4 global atomics (HBM-bound path).
// ---------------------------------------------------------------------------

typedef __bf16 bf16_t;
typedef bf16_t v16bf __attribute__((ext_vector_type(16)));
typedef float  v8f   __attribute__((ext_vector_type(8)));

static inline int cdiv(long a, long b) { return (int)((a + b - 1) / b); }

// ---- weight convert into fragment-ready WMMA B layout ---------------------
// o[((nt*KT + kt)*32 + lane)*16 + i] = bf16(W[k, col]), zero-padded k>=kvalid
//   col = nt*16 + (lane&15);  kb = (lane&16)?8:0
//   k   = kt*32 + (i<8 ? kb+i : 16+kb+(i-8))
__global__ void cvt_frag_bf16(const float* __restrict__ w, bf16_t* __restrict__ o,
                              int kvalid, int kt_count, int hout) {
    int tid = blockIdx.x * blockDim.x + threadIdx.x;  // one per (nt,kt,lane)
    int total = 8 * kt_count * 32;
    if (tid >= total) return;
    int lane = tid & 31;
    int kt = (tid >> 5) % kt_count;
    int nt = (tid >> 5) / kt_count;
    int col = nt * 16 + (lane & 15);
    int kb = (lane & 16) ? 8 : 0;
    bf16_t* op = o + (size_t)tid * 16;
#pragma unroll
    for (int i = 0; i < 16; ++i) {
        int k = kt * 32 + ((i < 8) ? (kb + i) : (16 + kb + (i - 8)));
        float v = (k < kvalid) ? w[(size_t)k * hout + col] : 0.0f;
        op[i] = (bf16_t)v;
    }
}

// ---- layer-1 z init: copy F cols into 32-wide zero-padded rows ------------
__global__ void pad_copyF(const float* __restrict__ x, float* __restrict__ z,
                          int n, int f) {
    int node = blockIdx.x * blockDim.x + threadIdx.x;
    if (node >= n) return;
    const float* xp = x + (size_t)node * f;
    float* zp = z + (size_t)node * 32;
    for (int c = 0; c < 32; ++c) zp[c] = (c < f) ? xp[c] : 0.0f;
}

// ---- edge scatter (first layer, F feats, z is N x 32 padded) --------------
__global__ void scatterF(const float* __restrict__ x, float* __restrict__ z,
                         const int* __restrict__ src, const int* __restrict__ dst,
                         int e, int f) {
    int edge = blockIdx.x * blockDim.x + threadIdx.x;
    if (edge >= e) return;
    int s = src[edge], d = dst[edge];
    for (int c = 0; c < f; ++c)
        atomicAdd(&z[(size_t)d * 32 + c], x[(size_t)s * f + c]);
}

// ---- edge scatter (H=128) -------------------------------------------------
__global__ void scatter128(const float* __restrict__ h, int ldh,
                           float* __restrict__ z,
                           const int* __restrict__ src, const int* __restrict__ dst,
                           int e) {
    long tid = (long)blockIdx.x * blockDim.x + threadIdx.x;
    int edge = (int)(tid >> 5);
    if (edge >= e) return;
    int c = ((int)tid & 31) * 4;
    int s = src[edge], d = dst[edge];
    float4 v = *(const float4*)(h + (size_t)s * ldh + c);
    float* zp = z + (size_t)d * 128 + c;
    atomicAdd(zp + 0, v.x);
    atomicAdd(zp + 1, v.y);
    atomicAdd(zp + 2, v.z);
    atomicAdd(zp + 3, v.w);
}

// ---- global add pool ------------------------------------------------------
__global__ void pool_add(const float* __restrict__ h, const int* __restrict__ batch,
                         float* __restrict__ g, int n) {
    long tid = (long)blockIdx.x * blockDim.x + threadIdx.x;
    int node = (int)(tid >> 5);
    if (node >= n) return;
    int c = ((int)tid & 31) * 4;
    int gi = batch[node];
    float4 v = *(const float4*)(h + (size_t)node * 128 + c);
    float* gp = g + (size_t)gi * 128 + c;
    atomicAdd(gp + 0, v.x);
    atomicAdd(gp + 1, v.y);
    atomicAdd(gp + 2, v.z);
    atomicAdd(gp + 3, v.w);
}

// ---- WMMA GEMM: Out = act(A @ W + bias), 128 output cols ------------------
// A: fp32 [M x LDA] (first KT*32 cols used; pad must be zero).
// Wfrag: fragment-ready bf16 (see cvt_frag_bf16). One wave per 16-row strip.
// Accumulator starts at inline 0; bias added in the epilogue. If DUAL, the
// result is also written to Out2 (ld=128) to seed the next scatter phase.
template <int KT, int LDA, int LDOUT, int RELU, int DUAL>
__global__ __launch_bounds__(128) void gemm_wmma(
    const float* __restrict__ A, const bf16_t* __restrict__ Wfrag,
    const float* __restrict__ bias, float* __restrict__ Out,
    float* __restrict__ Out2, int Mtiles) {
    extern __shared__ char smem[];
    bf16_t* sW = (bf16_t*)smem;                         // KT*4096 bf16
    float* sBias = (float*)(smem + (size_t)KT * 8192);  // 128 floats
    {
        const uint4* s4 = (const uint4*)Wfrag;
        uint4* d4 = (uint4*)sW;
        for (int i = threadIdx.x; i < KT * 512; i += 128) d4[i] = s4[i];
        sBias[threadIdx.x] = bias[threadIdx.x];
    }
    __syncthreads();

    int lane = threadIdx.x & 31;
    int strip = blockIdx.x * 4 + (threadIdx.x >> 5);
    if (strip >= Mtiles) return;

    int row0 = strip * 16;
    int m = lane & 15;
    int kb = (lane & 16) ? 8 : 0;

    // A fragments: two contiguous 8-float runs per k-tile -> 4x b128 loads
    v16bf a[KT];
#pragma unroll
    for (int kt = 0; kt < KT; ++kt) {
        const float* ap = A + (size_t)(row0 + m) * LDA + kt * 32 + kb;
        float4 f0 = *(const float4*)(ap + 0);
        float4 f1 = *(const float4*)(ap + 4);
        float4 f2 = *(const float4*)(ap + 16);
        float4 f3 = *(const float4*)(ap + 20);
        a[kt][0] = (bf16_t)f0.x;  a[kt][1] = (bf16_t)f0.y;
        a[kt][2] = (bf16_t)f0.z;  a[kt][3] = (bf16_t)f0.w;
        a[kt][4] = (bf16_t)f1.x;  a[kt][5] = (bf16_t)f1.y;
        a[kt][6] = (bf16_t)f1.z;  a[kt][7] = (bf16_t)f1.w;
        a[kt][8] = (bf16_t)f2.x;  a[kt][9] = (bf16_t)f2.y;
        a[kt][10] = (bf16_t)f2.z; a[kt][11] = (bf16_t)f2.w;
        a[kt][12] = (bf16_t)f3.x; a[kt][13] = (bf16_t)f3.y;
        a[kt][14] = (bf16_t)f3.z; a[kt][15] = (bf16_t)f3.w;
    }

    const v16bf* sWf = (const v16bf*)sW;
#pragma unroll
    for (int nt = 0; nt < 8; ++nt) {
        int col = nt * 16 + m;
        v8f acc = {};  // folds to inline-0 C operand of the first WMMA
#pragma unroll
        for (int kt = 0; kt < KT; ++kt) {
            v16bf b = sWf[(nt * KT + kt) * 32 + lane];  // 2x ds_load_b128
            acc = __builtin_amdgcn_wmma_f32_16x16x32_bf16(
                false, a[kt], false, b, (short)0, acc, false, false);
        }
        float bv = sBias[col];
        float* op = Out + (size_t)(row0 + kb) * LDOUT + col;
        float* op2 = DUAL ? (Out2 + (size_t)(row0 + kb) * 128 + col) : nullptr;
#pragma unroll
        for (int v = 0; v < 8; ++v) {
            float o = acc[v] + bv;
            if (RELU) o = fmaxf(o, 0.0f);
            op[(size_t)v * LDOUT] = o;  // immediate offsets (LDOUT constexpr)
            if (DUAL) op2[(size_t)v * 128] = o;
        }
    }
}

// ---- batch-norm (batch stats) + ReLU, in place ----------------------------
__global__ void bn_relu(float* __restrict__ z, const float* __restrict__ gam,
                        const float* __restrict__ bet, int g, int h) {
    int col = blockIdx.x;
    int t = threadIdx.x;
    float s = 0.0f, s2 = 0.0f;
    for (int r = t; r < g; r += blockDim.x) {
        float v = z[(size_t)r * h + col];
        s += v;
        s2 += v * v;
    }
    __shared__ float sh[256], sh2[256];
    sh[t] = s;
    sh2[t] = s2;
    __syncthreads();
    for (int st = 128; st > 0; st >>= 1) {
        if (t < st) {
            sh[t] += sh[t + st];
            sh2[t] += sh2[t + st];
        }
        __syncthreads();
    }
    float mean = sh[0] / (float)g;
    float var = sh2[0] / (float)g - mean * mean;
    float scale = rsqrtf(var + 1e-5f) * gam[col];
    float shift = bet[col];
    for (int r = t; r < g; r += blockDim.x) {
        float v = (z[(size_t)r * h + col] - mean) * scale + shift;
        z[(size_t)r * h + col] = fmaxf(v, 0.0f);
    }
}

// ---- tiny final linear [G x H] @ [H x C] ----------------------------------
__global__ void lin_small(const float* __restrict__ z, const float* __restrict__ w,
                          const float* __restrict__ b, float* __restrict__ out,
                          int g, int h, int c) {
    int i = blockIdx.x * blockDim.x + threadIdx.x;
    if (i >= g * c) return;
    int r = i / c, cc = i % c;
    float acc = b[cc];
    for (int k = 0; k < h; ++k) acc += z[(size_t)r * h + k] * w[(size_t)k * c + cc];
    out[i] = acc;
}

// ---------------------------------------------------------------------------
extern "C" void kernel_launch(void* const* d_in, const int* in_sizes, int n_in,
                              void* d_out, int out_size, void* d_ws, size_t ws_size,
                              hipStream_t stream) {
    const float* x = (const float*)d_in[0];
    const int* ei = (const int*)d_in[1];
    const int* batch = (const int*)d_in[3];
    const float* g1_w1 = (const float*)d_in[4];
    const float* g1_b1 = (const float*)d_in[5];
    const float* g1_w2 = (const float*)d_in[6];
    const float* g1_b2 = (const float*)d_in[7];
    const float* g2_w1 = (const float*)d_in[8];
    const float* g2_b1 = (const float*)d_in[9];
    const float* g2_w2 = (const float*)d_in[10];
    const float* g2_b2 = (const float*)d_in[11];
    const float* g3_w1 = (const float*)d_in[12];
    const float* g3_b1 = (const float*)d_in[13];
    const float* g3_w2 = (const float*)d_in[14];
    const float* g3_b2 = (const float*)d_in[15];
    const float* jk_w = (const float*)d_in[16];
    const float* jk_b = (const float*)d_in[17];
    const float* c1_w = (const float*)d_in[18];
    const float* c1_b = (const float*)d_in[19];
    const float* bn_g = (const float*)d_in[20];
    const float* bn_b = (const float*)d_in[21];
    const float* c2_w = (const float*)d_in[22];
    const float* c2_b = (const float*)d_in[23];

    const int E = in_sizes[2];      // edge_attr has E elements
    const int N = in_sizes[3];      // batch has N elements
    const int F = in_sizes[0] / N;  // 7
    const int H = in_sizes[5];      // 128
    const int C = in_sizes[23];     // 2
    const int G = out_size / C;     // 2048
    const int H3 = 3 * H;           // 384

    const int* src = ei;
    const int* dst = ei + E;

    // ---------------- workspace layout (256B aligned) ----------------
    size_t off = 0;
    auto alloc = [&](size_t bytes) {
        size_t r = off;
        off = (off + bytes + 255) & ~(size_t)255;
        return r;
    };
    char* ws = (char*)d_ws;
    float* zbuf  = (float*)(ws + alloc((size_t)N * H * 4));   // z for layers 2/3
    float* tbuf  = (float*)(ws + alloc((size_t)N * H * 4));   // MLP hidden / JK out
    float* hcat  = (float*)(ws + alloc((size_t)N * H3 * 4));  // [h1|h2|h3]
    float* z1    = (float*)(ws + alloc((size_t)N * 32 * 4));  // layer-1 z, padded ld=32
    float* gpool = (float*)(ws + alloc((size_t)G * H * 4));
    float* zc    = (float*)(ws + alloc((size_t)G * H * 4));
    bf16_t* wb_g1w1 = (bf16_t*)(ws + alloc((size_t)1  * 4096 * 2));
    bf16_t* wb_g1w2 = (bf16_t*)(ws + alloc((size_t)4  * 4096 * 2));
    bf16_t* wb_g2w1 = (bf16_t*)(ws + alloc((size_t)4  * 4096 * 2));
    bf16_t* wb_g2w2 = (bf16_t*)(ws + alloc((size_t)4  * 4096 * 2));
    bf16_t* wb_g3w1 = (bf16_t*)(ws + alloc((size_t)4  * 4096 * 2));
    bf16_t* wb_g3w2 = (bf16_t*)(ws + alloc((size_t)4  * 4096 * 2));
    bf16_t* wb_jk   = (bf16_t*)(ws + alloc((size_t)12 * 4096 * 2));
    bf16_t* wb_c1   = (bf16_t*)(ws + alloc((size_t)4  * 4096 * 2));
    (void)ws_size;

    const int Mtiles = (N + 15) / 16;  // 6250
    const int Gtiles = (G + 15) / 16;  // 128
    const int blkN = cdiv(Mtiles, 4);
    const int blkG = cdiv(Gtiles, 4);
    const size_t smem1  = (size_t)1  * 8192 + 512;
    const size_t smem4  = (size_t)4  * 8192 + 512;
    const size_t smem12 = (size_t)12 * 8192 + 512;

    // ---------------- weight conversion (fragment-ready bf16) ----------------
    cvt_frag_bf16<<<cdiv(8 * 1 * 32, 256), 256, 0, stream>>>(g1_w1, wb_g1w1, F, 1, H);
    cvt_frag_bf16<<<cdiv(8 * 4 * 32, 256), 256, 0, stream>>>(g1_w2, wb_g1w2, H, 4, H);
    cvt_frag_bf16<<<cdiv(8 * 4 * 32, 256), 256, 0, stream>>>(g2_w1, wb_g2w1, H, 4, H);
    cvt_frag_bf16<<<cdiv(8 * 4 * 32, 256), 256, 0, stream>>>(g2_w2, wb_g2w2, H, 4, H);
    cvt_frag_bf16<<<cdiv(8 * 4 * 32, 256), 256, 0, stream>>>(g3_w1, wb_g3w1, H, 4, H);
    cvt_frag_bf16<<<cdiv(8 * 4 * 32, 256), 256, 0, stream>>>(g3_w2, wb_g3w2, H, 4, H);
    cvt_frag_bf16<<<cdiv(8 * 12 * 32, 256), 256, 0, stream>>>(jk_w, wb_jk, H3, 12, H);
    cvt_frag_bf16<<<cdiv(8 * 4 * 32, 256), 256, 0, stream>>>(c1_w, wb_c1, H, 4, H);

    // ---------------- layer 1: z1 = x + scatter(x); h1 = relu(MLP(z1)) -------
    pad_copyF<<<cdiv(N, 256), 256, 0, stream>>>(x, z1, N, F);
    scatterF<<<cdiv(E, 256), 256, 0, stream>>>(x, z1, src, dst, E, F);
    gemm_wmma<1, 32, 128, 1, 0><<<blkN, 128, smem1, stream>>>(
        z1, wb_g1w1, g1_b1, tbuf, nullptr, Mtiles);
    // h1 -> hcat[:,0:128] AND zbuf (seed for layer-2 scatter)
    gemm_wmma<4, 128, 384, 1, 1><<<blkN, 128, smem4, stream>>>(
        tbuf, wb_g1w2, g1_b2, hcat + 0, zbuf, Mtiles);

    // ---------------- layer 2 -------------------------------------------------
    scatter128<<<cdiv((long)E * 32, 256), 256, 0, stream>>>(hcat + 0, H3, zbuf, src, dst, E);
    gemm_wmma<4, 128, 128, 1, 0><<<blkN, 128, smem4, stream>>>(
        zbuf, wb_g2w1, g2_b1, tbuf, nullptr, Mtiles);
    // h2 -> hcat[:,128:256] AND zbuf (seed for layer-3 scatter)
    gemm_wmma<4, 128, 384, 1, 1><<<blkN, 128, smem4, stream>>>(
        tbuf, wb_g2w2, g2_b2, hcat + H, zbuf, Mtiles);

    // ---------------- layer 3 -------------------------------------------------
    scatter128<<<cdiv((long)E * 32, 256), 256, 0, stream>>>(hcat + H, H3, zbuf, src, dst, E);
    gemm_wmma<4, 128, 128, 1, 0><<<blkN, 128, smem4, stream>>>(
        zbuf, wb_g3w1, g3_b1, tbuf, nullptr, Mtiles);
    gemm_wmma<4, 128, 384, 1, 0><<<blkN, 128, smem4, stream>>>(
        tbuf, wb_g3w2, g3_b2, hcat + 2 * H, nullptr, Mtiles);

    // ---------------- JK cat projection --------------------------------------
    gemm_wmma<12, 384, 128, 0, 0><<<blkN, 128, smem12, stream>>>(
        hcat, wb_jk, jk_b, tbuf, nullptr, Mtiles);

    // ---------------- global add pool ----------------------------------------
    hipMemsetAsync(gpool, 0, (size_t)G * H * 4, stream);
    pool_add<<<cdiv((long)N * 32, 256), 256, 0, stream>>>(tbuf, batch, gpool, N);

    // ---------------- classifier ---------------------------------------------
    gemm_wmma<4, 128, 128, 0, 0><<<blkG, 128, smem4, stream>>>(
        gpool, wb_c1, c1_b, zc, nullptr, Gtiles);
    bn_relu<<<H, 256, 0, stream>>>(zc, bn_g, bn_b, G, H);
    lin_small<<<cdiv(G * C, 256), 256, 0, stream>>>(zc, c2_w, c2_b, (float*)d_out,
                                                    G, H, C);
}
